// GAT_14181982012097
// MI455X (gfx1250) — matrix operand order repaired and checked by reading, not compiled, not run on hardware
//
#include <hip/hip_runtime.h>

// ---------------- problem constants (from reference) ----------------
constexpr int BATCH = 2;
constexpr int NN    = 2048;   // nodes
constexpr int FIN   = 128;    // input features
constexpr int HID   = 64;     // hidden per head
constexpr int NHD   = 4;      // heads
constexpr int HF    = NHD * HID;  // 256
constexpr int NOUT  = 64;
constexpr int NJ    = NN / 32;    // j-tiles per attention row block

typedef __attribute__((ext_vector_type(16))) __bf16 v16bf;
typedef __attribute__((ext_vector_type(8)))  __bf16 v8bf;
typedef __attribute__((ext_vector_type(8)))  float  v8f;
typedef unsigned int u32x4 __attribute__((ext_vector_type(4)));
typedef int          i32x4 __attribute__((ext_vector_type(4)));
typedef int          i32x8 __attribute__((ext_vector_type(8)));

#if defined(__has_builtin)
#if __has_builtin(__builtin_amdgcn_tensor_load_to_lds)
#define USE_TDM 1
#endif
#endif

__device__ __forceinline__ __bf16 f2bf(float f) {
  unsigned u = __builtin_bit_cast(unsigned, f);
  unsigned r = (u + 0x7FFFu + ((u >> 16) & 1u)) >> 16;
  unsigned short s = (unsigned short)r;
  return __builtin_bit_cast(__bf16, s);
}

__device__ __forceinline__ v8f wmma_bf16(v16bf a, v16bf b, v8f c) {
  return __builtin_amdgcn_wmma_f32_16x16x32_bf16(false, a, false, b, (short)0, c,
                                                 false, false);
}

#ifdef USE_TDM
// Issue a 2D TDM load: tile (tile0 x tile1) of bf16 (2-byte elems) from a
// row-major tensor with dim0 length `dim0` and dim0-stride `stride0` (elems),
// starting at `gsrc`, into LDS byte offset `lds_off`. D# per CDNA5 ISA 8.3/8.4.
__device__ __forceinline__ void tdm_load_tile(unsigned lds_off, const __bf16* gsrc,
                                              unsigned dim0, unsigned dim1,
                                              unsigned tile0, unsigned tile1,
                                              unsigned stride0) {
  unsigned long long ga = (unsigned long long)(size_t)gsrc;
  u32x4 g0 = {};
  g0[0] = 1u;                                           // count=1, user descriptor
  g0[1] = lds_off;                                      // lds_addr (bytes)
  g0[2] = (unsigned)(ga & 0xFFFFFFFFu);                 // global_addr[31:0]
  g0[3] = (unsigned)((ga >> 32) & 0x01FFFFFFu) | 0x80000000u;  // addr[56:32] | type=2
  i32x8 g1 = {};
  g1[0] = (int)(1u << 16);                              // data_size = 1 -> 2 bytes
  g1[1] = (int)((dim0 & 0xFFFFu) << 16);                // tensor_dim0[15:0]
  g1[2] = (int)(((dim0 >> 16) & 0xFFFFu) | ((dim1 & 0xFFFFu) << 16));  // dim0 hi | dim1 lo
  g1[3] = (int)(((dim1 >> 16) & 0xFFFFu) | (tile0 << 16));             // dim1 hi | tile_dim0
  g1[4] = (int)(tile1 & 0xFFFFu);                       // tile_dim1 (tile_dim2=0)
  g1[5] = (int)stride0;                                 // tensor_dim0_stride[31:0]
  g1[6] = 0;                                            // stride hi | dim1_stride lo (2D: unused)
  g1[7] = 0;
  i32x4 gz = {};
#if __clang_major__ >= 23
  i32x8 gz8 = {};
  __builtin_amdgcn_tensor_load_to_lds(g0, g1, gz, gz, gz8, 0);
#else
  __builtin_amdgcn_tensor_load_to_lds(g0, g1, gz, gz, 0);
#endif
}

__device__ __forceinline__ void tdm_wait() {
#if __has_builtin(__builtin_amdgcn_s_wait_tensorcnt)
  __builtin_amdgcn_s_wait_tensorcnt(0);
#else
  asm volatile("s_wait_tensorcnt 0x0" ::: "memory");
#endif
}
#endif  // USE_TDM

// ---------------- conversion / packing kernels ----------------
__global__ void k_cvt_x(const float* __restrict__ in, __bf16* __restrict__ out, int n) {
  int i = blockIdx.x * blockDim.x + threadIdx.x;
  if (i < n) out[i] = f2bf(in[i]);
}

// WT[h][o][k] = W[h][k][o]
__global__ void k_cvt_w(const float* __restrict__ W, __bf16* __restrict__ WT) {
  int i = blockIdx.x * blockDim.x + threadIdx.x;
  if (i >= NHD * HID * FIN) return;
  int hh = i / (HID * FIN);
  int r  = i % (HID * FIN);
  int o  = r / FIN;
  int k  = r % FIN;
  WT[i] = f2bf(W[((size_t)hh * FIN + k) * HID + o]);
}

// WT[o][k] = Wf[k][o]
__global__ void k_cvt_wf(const float* __restrict__ Wf, __bf16* __restrict__ WT) {
  int i = blockIdx.x * blockDim.x + threadIdx.x;
  if (i >= NOUT * HF) return;
  int o = i / HF;
  int k = i % HF;
  WT[i] = f2bf(Wf[(size_t)k * NOUT + o]);
}

// bit-pack adjacency: am[row][w] bit j  <=  adj[row][w*32+j] != 0
__global__ void k_adjpack(const int* __restrict__ adj, unsigned* __restrict__ am) {
  int i = blockIdx.x * blockDim.x + threadIdx.x;
  if (i >= BATCH * NN * (NN / 32)) return;
  int w = i % (NN / 32);
  size_t rowoff = (size_t)(i / (NN / 32)) * NN;
  const int* p = adj + rowoff + (size_t)w * 32;
  unsigned bits = 0;
  #pragma unroll
  for (int j = 0; j < 32; ++j) bits |= (p[j] != 0 ? 1u : 0u) << j;
  am[i] = bits;
}

// ---------------- fused GEMM: h = X @ W, plus s1 = h.a1, s2 = h.a2 ----------------
template <int KDIM, int NH>
__global__ __launch_bounds__(128) void k_gemm(const __bf16* __restrict__ X,
                                              const __bf16* __restrict__ WT,
                                              __bf16* __restrict__ HT,
                                              float* __restrict__ S1,
                                              float* __restrict__ S2,
                                              const float* __restrict__ A1,
                                              const float* __restrict__ A2) {
  __shared__ float lds[4][16][64];
  const int lane = threadIdx.x & 31;
  const int wave = threadIdx.x >> 5;
  const int tiles = NN / 64;
  const int bid  = blockIdx.x;
  const int tile = bid % tiles;
  const int bh   = bid / tiles;        // b*NH + h
  const int hh   = bh % NH;
  const int b    = bh / NH;
  const int m0   = tile * 64 + wave * 16;
  const int rlocal = lane & 15;
  const int mrow   = m0 + rlocal;

  const __bf16* xrow = X + ((size_t)b * NN + mrow) * KDIM;
  const int klo   = (lane < 16) ? 0 : 8;    // unused; A and B both load from bklo
  const int bklo  = (lane < 16) ? 0 : 16;   // B-layout K base
  const int mhalf = (lane < 16) ? 0 : 8;

  v8f c[4] = {};

  #pragma unroll 1
  for (int ks = 0; ks < KDIM / 32; ++ks) {
    const int kb = ks * 32;
    v8bf alo = *(const v8bf*)(xrow + kb + bklo);
    v8bf ahi = *(const v8bf*)(xrow + kb + bklo + 8);
    v16bf A;
    #pragma unroll
    for (int p = 0; p < 8; ++p) { A[p] = alo[p]; A[p + 8] = ahi[p]; }
    #pragma unroll
    for (int nt = 0; nt < 4; ++nt) {
      const __bf16* wcol =
          WT + ((size_t)hh * 64 + nt * 16 + rlocal) * KDIM + kb + bklo;
      v16bf Bv = *(const v16bf*)wcol;  // 16 contiguous K values
      c[nt] = wmma_bf16(A, Bv, c[nt]);
    }
  }

  // store transposed bf16 H (attention V operand) + stage f32 tile for s1/s2
  #pragma unroll
  for (int nt = 0; nt < 4; ++nt) {
    #pragma unroll
    for (int r = 0; r < 8; ++r) {
      float v = c[nt][r];
      int m = m0 + r + mhalf;
      int o = nt * 16 + rlocal;
      HT[((size_t)bh * 64 + o) * NN + m] = f2bf(v);
      lds[wave][r + mhalf][o] = v;
    }
  }
  __syncthreads();
  if (lane < 16) {
    int m = m0 + lane;
    float s1v = 0.0f, s2v = 0.0f;
    #pragma unroll
    for (int o = 0; o < 64; ++o) {
      float hv = lds[wave][lane][o];
      s1v += hv * A1[hh * 64 + o];
      s2v += hv * A2[hh * 64 + o];
    }
    S1[(size_t)bh * NN + m] = s1v;
    S2[(size_t)bh * NN + m] = s2v;
  }
}

// ---------------- layer-1 fused masked-softmax attention (flash style) ----------------
// All 8 waves of a block share one (b,head): V tiles are TDM-staged in LDS once
// per block (double-buffered), overlapping the DMA with softmax + WMMA.
__global__ __launch_bounds__(256) void k_attn1(const float* __restrict__ S1,
                                               const float* __restrict__ S2,
                                               const __bf16* __restrict__ HT,
                                               const unsigned* __restrict__ ADJM,
                                               __bf16* __restrict__ CATB) {
#ifdef USE_TDM
  __shared__ __align__(128) __bf16 vstage[2][64 * 32];  // [buf][o*32 + j]
#endif
  const int lane = threadIdx.x & 31;
  const int wave = threadIdx.x >> 5;
  const int t  = blockIdx.x * 8 + wave;      // m-tile over B*NHD*(NN/16)
  const int mt = t % (NN / 16);
  const int bh = t / (NN / 16);              // uniform per block (8 | NN/16)
  const int b  = bh / NHD;
  const int hh = bh % NHD;
  const int m0 = mt * 16;
  const int rlocal = lane & 15;
  const int mrow   = m0 + rlocal;
  const int klo    = (lane < 16) ? 0 : 8;
  const int bklo   = (lane < 16) ? 0 : 16;
  const int mhalf  = (lane < 16) ? 0 : 8;

  const float s1row = S1[(size_t)bh * NN + mrow];
  const float* s2p  = S2 + (size_t)bh * NN;
  const unsigned* amrow = ADJM + ((size_t)b * NN + mrow) * (NN / 32);
  const __bf16* vbase   = HT + (size_t)bh * 64 * NN;

  float rm = -1.0e30f, rl = 0.0f;
  v8f acc[4] = {};

#ifdef USE_TDM
  const unsigned lds0 = (unsigned)(unsigned long long)(size_t)&vstage[0][0];
  if (wave == 0) tdm_load_tile(lds0, vbase, NN, 64, 32, 64, NN);
#endif

  #pragma unroll 1
  for (int jt = 0; jt < NJ; ++jt) {
    const int jb = jt * 32;
#ifdef USE_TDM
    if (wave == 0) tdm_wait();               // tile jt landed in LDS
    __syncthreads();                         // publish tile jt / retire other buf
    if (wave == 0 && jt + 1 < NJ)
      tdm_load_tile(lds0 + (unsigned)(((jt + 1) & 1) * (64 * 32 * 2)),
                    vbase + (size_t)(jt + 1) * 32, NN, 64, 32, 64, NN);
    const __bf16* vtile = &vstage[jt & 1][0];
#endif
    const unsigned mword = amrow[jt];
    __builtin_prefetch(amrow + jt + 1, 0, 0);  // global_prefetch_b8 on mask stream

    v8f s2lo = *(const v8f*)(s2p + jb + bklo);
    v8f s2hi = *(const v8f*)(s2p + jb + bklo + 8);

    float e[16];
    float tmax = -1.0e30f;
    #pragma unroll
    for (int i = 0; i < 16; ++i) {
      int k = bklo + i;   // matches A-operand K layout (contiguous)
      float ev = s1row + ((i < 8) ? s2lo[i] : s2hi[i - 8]);
      ev = (ev >= 0.0f) ? ev : 0.2f * ev;      // LeakyReLU(0.2)
      ev = ((mword >> k) & 1u) ? ev : -1.0e9f; // adjacency mask
      e[i] = ev;
      tmax = fmaxf(tmax, ev);
    }
    tmax = fmaxf(tmax, __shfl_xor(tmax, 16, 32));
    float mnew = fmaxf(rm, tmax);
    float sc   = __expf(rm - mnew);
    float ls   = 0.0f;
    v16bf P;
    #pragma unroll
    for (int i = 0; i < 16; ++i) {
      float pv = __expf(e[i] - mnew);
      ls += pv;
      P[i] = f2bf(pv);
    }
    ls += __shfl_xor(ls, 16, 32);
    rl = rl * sc + ls;
    rm = mnew;

    // rescale accumulators per row (broadcast the owning lane's factor)
    #pragma unroll
    for (int r = 0; r < 8; ++r) {
      float scr = __shfl(sc, r + mhalf, 32);
      acc[0][r] = acc[0][r] * scr;
      acc[1][r] = acc[1][r] * scr;
      acc[2][r] = acc[2][r] * scr;
      acc[3][r] = acc[3][r] * scr;
    }
    #pragma unroll
    for (int nt = 0; nt < 4; ++nt) {
#ifdef USE_TDM
      v16bf Bv = *(const v16bf*)(vtile + (nt * 16 + rlocal) * 32 + bklo);
#else
      const __bf16* vcol = vbase + ((size_t)(nt * 16 + rlocal)) * NN + jb + bklo;
      v16bf Bv = *(const v16bf*)vcol;
#endif
      acc[nt] = wmma_bf16(P, Bv, acc[nt]);
    }
  }

  // normalize, ELU, store concatenated bf16 features for layer 2
  #pragma unroll
  for (int r = 0; r < 8; ++r) {
    float inv = 1.0f / __shfl(rl, r + mhalf, 32);
    int m = m0 + r + mhalf;
    #pragma unroll
    for (int nt = 0; nt < 4; ++nt) {
      float v = acc[nt][r] * inv;
      v = (v > 0.0f) ? v : (__expf(v) - 1.0f);
      CATB[((size_t)b * NN + m) * HF + hh * HID + nt * 16 + rlocal] = f2bf(v);
    }
  }
}

// ---------------- layer-2 attention + ELU + log_softmax ----------------
__global__ __launch_bounds__(256) void k_attn2(const float* __restrict__ S1F,
                                               const float* __restrict__ S2F,
                                               const __bf16* __restrict__ H2T,
                                               const unsigned* __restrict__ ADJM,
                                               float* __restrict__ OUTP) {
#ifdef USE_TDM
  __shared__ __align__(128) __bf16 vstage[2][64 * 32];
#endif
  const int lane = threadIdx.x & 31;
  const int wave = threadIdx.x >> 5;
  const int t  = blockIdx.x * 8 + wave;      // m-tile over B*(NN/16)
  const int mt = t % (NN / 16);
  const int b  = t / (NN / 16);              // uniform per block
  const int m0 = mt * 16;
  const int rlocal = lane & 15;
  const int mrow   = m0 + rlocal;
  const int klo    = (lane < 16) ? 0 : 8;
  const int bklo   = (lane < 16) ? 0 : 16;
  const int mhalf  = (lane < 16) ? 0 : 8;

  const float s1row = S1F[(size_t)b * NN + mrow];
  const float* s2p  = S2F + (size_t)b * NN;
  const unsigned* amrow = ADJM + ((size_t)b * NN + mrow) * (NN / 32);
  const __bf16* vbase   = H2T + (size_t)b * 64 * NN;

  float rm = -1.0e30f, rl = 0.0f;
  v8f acc[4] = {};

#ifdef USE_TDM
  const unsigned lds0 = (unsigned)(unsigned long long)(size_t)&vstage[0][0];
  if (wave == 0) tdm_load_tile(lds0, vbase, NN, 64, 32, 64, NN);
#endif

  #pragma unroll 1
  for (int jt = 0; jt < NJ; ++jt) {
    const int jb = jt * 32;
#ifdef USE_TDM
    if (wave == 0) tdm_wait();
    __syncthreads();
    if (wave == 0 && jt + 1 < NJ)
      tdm_load_tile(lds0 + (unsigned)(((jt + 1) & 1) * (64 * 32 * 2)),
                    vbase + (size_t)(jt + 1) * 32, NN, 64, 32, 64, NN);
    const __bf16* vtile = &vstage[jt & 1][0];
#endif
    const unsigned mword = amrow[jt];
    __builtin_prefetch(amrow + jt + 1, 0, 0);

    v8f s2lo = *(const v8f*)(s2p + jb + bklo);
    v8f s2hi = *(const v8f*)(s2p + jb + bklo + 8);

    float e[16];
    float tmax = -1.0e30f;
    #pragma unroll
    for (int i = 0; i < 16; ++i) {
      int k = bklo + i;
      float ev = s1row + ((i < 8) ? s2lo[i] : s2hi[i - 8]);
      ev = (ev >= 0.0f) ? ev : 0.2f * ev;
      ev = ((mword >> k) & 1u) ? ev : -1.0e9f;
      e[i] = ev;
      tmax = fmaxf(tmax, ev);
    }
    tmax = fmaxf(tmax, __shfl_xor(tmax, 16, 32));
    float mnew = fmaxf(rm, tmax);
    float sc   = __expf(rm - mnew);
    float ls   = 0.0f;
    v16bf P;
    #pragma unroll
    for (int i = 0; i < 16; ++i) {
      float pv = __expf(e[i] - mnew);
      ls += pv;
      P[i] = f2bf(pv);
    }
    ls += __shfl_xor(ls, 16, 32);
    rl = rl * sc + ls;
    rm = mnew;

    #pragma unroll
    for (int r = 0; r < 8; ++r) {
      float scr = __shfl(sc, r + mhalf, 32);
      acc[0][r] = acc[0][r] * scr;
      acc[1][r] = acc[1][r] * scr;
      acc[2][r] = acc[2][r] * scr;
      acc[3][r] = acc[3][r] * scr;
    }
    #pragma unroll
    for (int nt = 0; nt < 4; ++nt) {
#ifdef USE_TDM
      v16bf Bv = *(const v16bf*)(vtile + (nt * 16 + rlocal) * 32 + bklo);
#else
      const __bf16* vcol = vbase + ((size_t)(nt * 16 + rlocal)) * NN + jb + bklo;
      v16bf Bv = *(const v16bf*)vcol;
#endif
      acc[nt] = wmma_bf16(P, Bv, acc[nt]);
    }
  }

  // normalize + ELU + row-wise log_softmax over the 64 outputs
  #pragma unroll
  for (int r = 0; r < 8; ++r) {
    float inv = 1.0f / __shfl(rl, r + mhalf, 32);
    float y[4];
    float xm = -1.0e30f;
    #pragma unroll
    for (int nt = 0; nt < 4; ++nt) {
      float v = acc[nt][r] * inv;
      v = (v > 0.0f) ? v : (__expf(v) - 1.0f);
      y[nt] = v;
      xm = fmaxf(xm, v);
    }
    xm = fmaxf(xm, __shfl_xor(xm, 1, 32));
    xm = fmaxf(xm, __shfl_xor(xm, 2, 32));
    xm = fmaxf(xm, __shfl_xor(xm, 4, 32));
    xm = fmaxf(xm, __shfl_xor(xm, 8, 32));
    float ss = 0.0f;
    #pragma unroll
    for (int nt = 0; nt < 4; ++nt) ss += __expf(y[nt] - xm);
    ss += __shfl_xor(ss, 1, 32);
    ss += __shfl_xor(ss, 2, 32);
    ss += __shfl_xor(ss, 4, 32);
    ss += __shfl_xor(ss, 8, 32);
    float lg = __logf(ss) + xm;
    int m = m0 + r + mhalf;
    #pragma unroll
    for (int nt = 0; nt < 4; ++nt)
      OUTP[((size_t)b * NN + m) * NOUT + nt * 16 + rlocal] = y[nt] - lg;
  }
}

// ---------------- host-side launch ----------------
extern "C" void kernel_launch(void* const* d_in, const int* in_sizes, int n_in,
                              void* d_out, int out_size, void* d_ws, size_t ws_size,
                              hipStream_t stream) {
  const float* nodes = (const float*)d_in[0];
  const int*   adj   = (const int*)d_in[1];
  const float* W     = (const float*)d_in[2];
  const float* a1    = (const float*)d_in[3];
  const float* a2    = (const float*)d_in[4];
  const float* Wf    = (const float*)d_in[5];
  const float* a1f   = (const float*)d_in[6];
  const float* a2f   = (const float*)d_in[7];
  float* outp = (float*)d_out;

  char* ws = (char*)d_ws;
  size_t off = 0;
  auto alloc = [&](size_t bytes) -> char* {
    char* p = ws + off;
    off = (off + bytes + 255) & ~(size_t)255;
    return p;
  };

  __bf16* xb   = (__bf16*)alloc((size_t)BATCH * NN * FIN * 2);       // bf16 nodes
  __bf16* WbT  = (__bf16*)alloc((size_t)NHD * HID * FIN * 2);        // W^T per head
  __bf16* WfbT = (__bf16*)alloc((size_t)NOUT * HF * 2);              // Wf^T
  unsigned* adjm = (unsigned*)alloc((size_t)BATCH * NN * (NN / 32) * 4);
  __bf16* h1T  = (__bf16*)alloc((size_t)BATCH * NHD * HID * NN * 2); // layer-1 V
  float* s1    = (float*)alloc((size_t)BATCH * NHD * NN * 4);
  float* s2    = (float*)alloc((size_t)BATCH * NHD * NN * 4);
  __bf16* catb = (__bf16*)alloc((size_t)BATCH * NN * HF * 2);        // concat feats
  __bf16* h2T  = (__bf16*)alloc((size_t)BATCH * NOUT * NN * 2);      // layer-2 V
  float* s1f   = (float*)alloc((size_t)BATCH * NN * 4);
  float* s2f   = (float*)alloc((size_t)BATCH * NN * 4);

  // 1) precision conversion + adjacency bit-packing
  {
    int n = BATCH * NN * FIN;
    k_cvt_x<<<(n + 255) / 256, 256, 0, stream>>>(nodes, xb, n);
  }
  k_cvt_w<<<(NHD * HID * FIN + 255) / 256, 256, 0, stream>>>(W, WbT);
  k_cvt_wf<<<(NOUT * HF + 255) / 256, 256, 0, stream>>>(Wf, WfbT);
  k_adjpack<<<(BATCH * NN * (NN / 32) + 255) / 256, 256, 0, stream>>>(adj, adjm);

  // 2) layer-1 feature GEMM (+s1/s2)
  k_gemm<FIN, NHD><<<BATCH * NHD * (NN / 64), 128, 0, stream>>>(
      xb, WbT, h1T, s1, s2, a1, a2);

  // 3) layer-1 fused masked-softmax attention + ELU -> cat (bf16)
  k_attn1<<<BATCH * NHD * (NN / 16) / 8, 256, 0, stream>>>(s1, s2, h1T, adjm, catb);

  // 4) layer-2 GEMM (+s1f/s2f)
  k_gemm<HF, 1><<<BATCH * (NN / 64), 128, 0, stream>>>(
      catb, WfbT, h2T, s1f, s2f, a1f, a2f);

  // 5) layer-2 attention + ELU + log_softmax -> output
  k_attn2<<<BATCH * (NN / 16) / 8, 256, 0, stream>>>(s1f, s2f, h2T, adjm, outp);
}
